// TopKRouter_60644938219690
// MI455X (gfx1250) — matrix operand order repaired and checked by reading, not compiled, not run on hardware
//
#include <hip/hip_runtime.h>
#include <hip/hip_bf16.h>
#include <math.h>

typedef __attribute__((ext_vector_type(16))) __bf16        v16bf;
typedef __attribute__((ext_vector_type(8)))  __bf16        v8bf;
typedef __attribute__((ext_vector_type(4)))  __bf16        v4bf;
typedef __attribute__((ext_vector_type(8)))  float         v8f;
typedef __attribute__((ext_vector_type(8)))  unsigned int  v8u;

#define HDIM        4096
#define NEXP        256
#define TOKS_PER_WG 64
#define KCHUNK      32
#define NCHUNK      (HDIM / KCHUNK)
#define SSTRIDE     261    // floats; 261 mod 64 = 5 (odd) -> conflict-free column scans
#define SCALE_F     2.5f
#define EPS_F       1e-9f

#define WS_ACC_FLOATS 512                         // cnt[256] + prob[256]
#define WS_W_OFFSET   2048                        // bytes
#define WS_W_BYTES    ((size_t)NEXP * HDIM * 2)   // 2 MB per array
#define WS_NEEDED     (WS_W_OFFSET + 2 * WS_W_BYTES)

// ---------------------------------------------------------------------------
// zero the 512-float accumulator region of d_ws (counts[256] + probsum[256])
// ---------------------------------------------------------------------------
__global__ void router_init_kernel(float* acc) {
    acc[threadIdx.x] = 0.0f;
}

// ---------------------------------------------------------------------------
// one-time split of W (f32) into hi/lo bf16 arrays (row-major, same indexing)
// ---------------------------------------------------------------------------
__global__ void wsplit_kernel(const float* __restrict__ W,
                              unsigned short* __restrict__ whi,
                              unsigned short* __restrict__ wlo)
{
    const int i = (blockIdx.x * 256 + threadIdx.x);      // float4 index
    float4 v = ((const float4*)W)[i];
    float f[4] = {v.x, v.y, v.z, v.w};
    v4bf h4, l4;
#pragma unroll
    for (int j = 0; j < 4; ++j) {
        __bf16 h = (__bf16)f[j];
        h4[j] = h;
        l4[j] = (__bf16)(f[j] - (float)h);
    }
    *(v4bf*)(whi + 4 * (size_t)i) = h4;
    *(v4bf*)(wlo + 4 * (size_t)i) = l4;
}

// ---------------------------------------------------------------------------
// Main kernel: bf16x3 WMMA router GEMM + sigmoid + group top-4 + top-8.
// One 256-thread workgroup handles 64 tokens x 256 experts.
// A chunk is split hi/lo ONCE cooperatively into fragment-ready swizzled LDS.
// ---------------------------------------------------------------------------
template <bool PRESPLIT>
__global__ __launch_bounds__(256)
void router_main_kernel(const float* __restrict__ tokens,
                        const float* __restrict__ W,
                        const unsigned short* __restrict__ whi,
                        const unsigned short* __restrict__ wlo,
                        float* __restrict__ out_idx,   // [N,8] indices as float
                        float* __restrict__ out_w,     // [N,8] weights
                        float* __restrict__ cnt,       // [256]
                        float* __restrict__ prob)      // [256]
{
    // fragment-ready A: [buf][ (mt*2+half)*32 + lane ] -> 8 bf16 (16 B)
    __shared__ uint4 aHI[2][256];                      // 8 KB
    __shared__ uint4 aLO[2][256];                      // 8 KB
    __shared__ float sLDS[TOKS_PER_WG * SSTRIDE];      // 66816 B
    __shared__ float rowinv[TOKS_PER_WG];

    const int tid   = threadIdx.x;
    const int wave  = tid >> 5;
    const int lane  = tid & 31;
    const int l16   = lane & 15;
    const bool lohi = lane >= 16;
    const int koff  = lohi ? 16 : 0;
    const int tok0  = blockIdx.x * TOKS_PER_WG;

    // ---- staging-role constants: this thread's 8 consecutive K values of one
    //      token row land in exactly one 16-byte fragment-half of one lane ----
    const int r_s   = tid >> 2;                 // token row 0..63
    const int q_s   = tid & 3;                  // K quarter 0..3
    const int aslot = (((r_s >> 4) * 2 + (q_s & 1)) * 32)    // [mt][half]
                      + (r_s & 15) + ((q_s >> 1) ? 16 : 0);  // dest lane
    const float* asrc = tokens + (size_t)(tok0 + r_s) * HDIM + q_s * 8;

    // ---- accumulators: [expert tile 0..1][token tile 0..3] ----
    v8f acc[2][4];
#pragma unroll
    for (int j = 0; j < 2; ++j)
#pragma unroll
        for (int mt = 0; mt < 4; ++mt)
            acc[j][mt] = (v8f){0.f,0.f,0.f,0.f,0.f,0.f,0.f,0.f};

    // prefetch chunk 0
    float4 pv0 = *(const float4*)(asrc);
    float4 pv1 = *(const float4*)(asrc + 4);

    for (int kc = 0; kc < NCHUNK; ++kc) {
        const int kbase = kc * KCHUNK;
        const int buf   = kc & 1;

        // ---- convert + store prefetched A values (hi/lo split, swizzled) ----
        {
            float f[8] = {pv0.x, pv0.y, pv0.z, pv0.w, pv1.x, pv1.y, pv1.z, pv1.w};
            v8bf h8, l8;
#pragma unroll
            for (int i = 0; i < 8; ++i) {
                __bf16 h = (__bf16)f[i];
                h8[i] = h;
                l8[i] = (__bf16)(f[i] - (float)h);
            }
            aHI[buf][aslot] = __builtin_bit_cast(uint4, h8);
            aLO[buf][aslot] = __builtin_bit_cast(uint4, l8);
        }
        // prefetch next chunk
        if (kc + 1 < NCHUNK) {
            pv0 = *(const float4*)(asrc + (kc + 1) * KCHUNK);
            pv1 = *(const float4*)(asrc + (kc + 1) * KCHUNK + 4);
        }
        __syncthreads();

        // ---- B fragments (lane = expert col; 16 contiguous K values) ----
        v16bf bhi[2], blo[2];
#pragma unroll
        for (int j = 0; j < 2; ++j) {
            const int e = (wave * 2 + j) * 16 + l16;
            if (PRESPLIT) {
                const uint4* ph = (const uint4*)(whi + (size_t)e * HDIM + kbase + koff);
                const uint4* pl = (const uint4*)(wlo + (size_t)e * HDIM + kbase + koff);
                uint4 h0 = ph[0], h1 = ph[1];
                uint4 l0 = pl[0], l1 = pl[1];
                bhi[j] = __builtin_bit_cast(v16bf,
                    (v8u){h0.x, h0.y, h0.z, h0.w, h1.x, h1.y, h1.z, h1.w});
                blo[j] = __builtin_bit_cast(v16bf,
                    (v8u){l0.x, l0.y, l0.z, l0.w, l1.x, l1.y, l1.z, l1.w});
            } else {
                const float* wr = W + (size_t)e * HDIM + kbase + koff;
#pragma unroll
                for (int qq = 0; qq < 4; ++qq) {
                    float4 v = *(const float4*)(wr + qq * 4);
                    float f[4] = {v.x, v.y, v.z, v.w};
#pragma unroll
                    for (int i = 0; i < 4; ++i) {
                        __bf16 h = (__bf16)f[i];
                        bhi[j][qq * 4 + i] = h;
                        blo[j][qq * 4 + i] = (__bf16)(f[i] - (float)h);
                    }
                }
            }
        }

        // ---- token tiles: load fragment-ready A, then bf16x3 WMMA ----
#pragma unroll
        for (int mt = 0; mt < 4; ++mt) {
            uint4 h0 = aHI[buf][(mt * 2 + 0) * 32 + lane];
            uint4 h1 = aHI[buf][(mt * 2 + 1) * 32 + lane];
            uint4 l0 = aLO[buf][(mt * 2 + 0) * 32 + lane];
            uint4 l1 = aLO[buf][(mt * 2 + 1) * 32 + lane];
            v16bf ahi = __builtin_bit_cast(v16bf,
                (v8u){h0.x, h0.y, h0.z, h0.w, h1.x, h1.y, h1.z, h1.w});
            v16bf alo = __builtin_bit_cast(v16bf,
                (v8u){l0.x, l0.y, l0.z, l0.w, l1.x, l1.y, l1.z, l1.w});
#pragma unroll
            for (int j = 0; j < 2; ++j) {
                acc[j][mt] = __builtin_amdgcn_wmma_f32_16x16x32_bf16(
                    false, ahi, false, bhi[j], (short)0, acc[j][mt], false, false);
                acc[j][mt] = __builtin_amdgcn_wmma_f32_16x16x32_bf16(
                    false, ahi, false, blo[j], (short)0, acc[j][mt], false, false);
                acc[j][mt] = __builtin_amdgcn_wmma_f32_16x16x32_bf16(
                    false, alo, false, bhi[j], (short)0, acc[j][mt], false, false);
            }
        }
        // no trailing barrier: next iteration stages into the other buffer
    }
    __syncthreads();

    // ---- sigmoid + scatter scores to LDS (C layout: VGPR r = row r / r+8) ----
#pragma unroll
    for (int j = 0; j < 2; ++j) {
        const int e = (wave * 2 + j) * 16 + l16;
#pragma unroll
        for (int mt = 0; mt < 4; ++mt) {
#pragma unroll
            for (int r = 0; r < 8; ++r) {
                const int tok = mt * 16 + r + (lohi ? 8 : 0);
                float x = acc[j][mt][r];
                sLDS[tok * SSTRIDE + e] = 1.0f / (1.0f + __expf(-x));
            }
        }
    }
    __syncthreads();

    // ---- routing: threads 0..63, one token each ----
    if (tid < TOKS_PER_WG) {
        const float* srow = &sLDS[tid * SSTRIDE];
        float gmax[8];
#pragma unroll
        for (int g = 0; g < 8; ++g) gmax[g] = -1.0f;
        float rs = 0.0f;
        for (int e = 0; e < NEXP; ++e) {
            float s = srow[e];
            rs += s;
            gmax[e >> 5] = fmaxf(gmax[e >> 5], s);
        }
        // top-4 groups (stable compare-swap chain; strict > matches jax ties)
        float gv[4] = {-1.f, -1.f, -1.f, -1.f};
        int   gi[4] = {0, 0, 0, 0};
#pragma unroll
        for (int g = 0; g < 8; ++g) {
            float cur = gmax[g]; int ci = g;
#pragma unroll
            for (int j = 0; j < 4; ++j) {
                bool sw = cur > gv[j];
                float tv = gv[j]; int ti = gi[j];
                if (sw) { gv[j] = cur; gi[j] = ci; cur = tv; ci = ti; }
            }
        }
        unsigned mask = 0;
#pragma unroll
        for (int j = 0; j < 4; ++j) mask |= 1u << gi[j];

        // stable top-8 over group-allowed experts
        float tw[8]; int ti[8];
#pragma unroll
        for (int j = 0; j < 8; ++j) { tw[j] = -1.0f; ti[j] = 0; }
        for (int e = 0; e < NEXP; ++e) {
            bool allowed = (mask >> (e >> 5)) & 1u;
            float cur = allowed ? srow[e] : -1.0f;
            int ci = e;
#pragma unroll
            for (int j = 0; j < 8; ++j) {
                bool sw = cur > tw[j];
                float tv = tw[j]; int tx = ti[j];
                if (sw) { tw[j] = cur; ti[j] = ci; cur = tv; ci = tx; }
            }
        }
        float denom = 0.0f;
#pragma unroll
        for (int j = 0; j < 8; ++j) denom += tw[j];
        const float sc = SCALE_F / fmaxf(denom, EPS_F);

        const size_t ob = (size_t)(tok0 + tid) * 8;
#pragma unroll
        for (int j = 0; j < 8; ++j) {
            out_idx[ob + j] = (float)ti[j];
            out_w[ob + j]   = tw[j] * sc;
            atomicAdd(&cnt[ti[j]], 1.0f);
        }
        rowinv[tid] = 1.0f / fmaxf(rs, EPS_F);
    }
    __syncthreads();

    // ---- aux-loss partial: thread per expert, sum normed scores over tokens ----
    {
        const int e = tid;
        float p = 0.0f;
        for (int t = 0; t < TOKS_PER_WG; ++t)
            p += sLDS[t * SSTRIDE + e] * rowinv[t];
        atomicAdd(&prob[e], p);
    }
}

// ---------------------------------------------------------------------------
// Finale: aux_loss = sum_e (cnt[e]/(N*K)) * (prob[e]/N) * E
// ---------------------------------------------------------------------------
__global__ void router_aux_kernel(const float* __restrict__ cnt,
                                  const float* __restrict__ prob,
                                  float* __restrict__ out_aux, float invNK, float invN)
{
    __shared__ float red[256];
    const int e = threadIdx.x;
    red[e] = (cnt[e] * invNK) * (prob[e] * invN);
    __syncthreads();
#pragma unroll
    for (int s = 128; s > 0; s >>= 1) {
        if (e < s) red[e] += red[e + s];
        __syncthreads();
    }
    if (e == 0) out_aux[0] = red[0] * (float)NEXP;
}

// ---------------------------------------------------------------------------
extern "C" void kernel_launch(void* const* d_in, const int* in_sizes, int n_in,
                              void* d_out, int out_size, void* d_ws, size_t ws_size,
                              hipStream_t stream)
{
    const float* tokens = (const float*)d_in[0];
    const float* W      = (const float*)d_in[1];
    const int N = in_sizes[0] / HDIM;            // 16384

    float* out     = (float*)d_out;
    float* out_idx = out;
    float* out_w   = out + (size_t)N * 8;
    float* out_aux = out + (size_t)2 * N * 8;

    float* cnt  = (float*)d_ws;                  // [256]
    float* prob = cnt + NEXP;                    // [256]
    unsigned short* whi = (unsigned short*)((char*)d_ws + WS_W_OFFSET);
    unsigned short* wlo = (unsigned short*)((char*)d_ws + WS_W_OFFSET + WS_W_BYTES);

    const bool presplit = ws_size >= WS_NEEDED;

    router_init_kernel<<<1, WS_ACC_FLOATS, 0, stream>>>(cnt);

    if (presplit) {
        // 256*4096 f32 = 262144 float4s -> 1024 blocks of 256
        wsplit_kernel<<<(NEXP * HDIM / 4) / 256, 256, 0, stream>>>(W, whi, wlo);
        router_main_kernel<true><<<N / TOKS_PER_WG, 256, 0, stream>>>(
            tokens, W, whi, wlo, out_idx, out_w, cnt, prob);
    } else {
        router_main_kernel<false><<<N / TOKS_PER_WG, 256, 0, stream>>>(
            tokens, W, whi, wlo, out_idx, out_w, cnt, prob);
    }

    router_aux_kernel<<<1, 256, 0, stream>>>(
        cnt, prob, out_aux, 1.0f / ((float)N * 8.0f), 1.0f / (float)N);
}
